// Attention_39436389712198
// MI455X (gfx1250) — matrix operand order repaired and checked by reading, not compiled, not run on hardware
//
#include <hip/hip_runtime.h>
#include <hip/hip_bf16.h>

// Shapes from the reference
#define B_    32
#define T_    2048
#define F_    512
#define H_    128
#define C_    32
#define KC_   31
#define KAUG  544          // F_ + C_  (augmented GEMM K: Wk columns ++ Ww columns)
#define NKSTEP 17          // KAUG / 32

typedef __attribute__((ext_vector_type(16))) _Float16 v16h;
typedef __attribute__((ext_vector_type(8)))  _Float16 v8h;
typedef __attribute__((ext_vector_type(8)))  float    v8f;

// ---------------------------------------------------------------------------
// Kernel 1: build augmented, transposed, f16 B-matrix  BT[k][h], k in [0,544)
//   k <  512 : Wk[h][k]
//   k >= 512 : Ww[h][k-512]
// Stored k-major so the hot kernel's LDS B-slab fill is a coalesced row copy.
// ---------------------------------------------------------------------------
__global__ void build_bt(const float* __restrict__ Wk,
                         const float* __restrict__ Ww,
                         _Float16* __restrict__ BT) {
    int o = blockIdx.x * 256 + threadIdx.x;
    if (o >= KAUG * H_) return;
    int k = o / H_, h = o % H_;
    float v = (k < F_) ? Wk[h * F_ + k] : Ww[h * C_ + (k - F_)];
    BT[k * H_ + h] = (_Float16)v;
}

// ---------------------------------------------------------------------------
// Kernel 2: q = query @ Wq^T + bias   [32,128] — tiny, fp32 for accuracy.
// ---------------------------------------------------------------------------
__global__ void qproj(const float* __restrict__ query,
                      const float* __restrict__ Wq,
                      const float* __restrict__ bias,
                      float* __restrict__ q) {
    int o = blockIdx.x * 256 + threadIdx.x;   // 4096 = B_*H_
    int b = o >> 7, h = o & 127;
    const float* qr = query + (size_t)b * (2 * F_);
    const float* wr = Wq    + (size_t)h * (2 * F_);
    float acc = 0.f;
    for (int i = 0; i < 2 * F_; ++i) acc += qr[i] * wr[i];
    q[o] = acc + bias[h];
}

// ---------------------------------------------------------------------------
// Kernel 3 (hot): fused  scores[b,t] = Ws . tanh( kv@Wk^T + conv_feat@Ww^T
//                                                 + q[b] + bias )
// One WG = (b, 128-row T tile). 8 waves; each wave owns 16 rows x full H=128
// (8 WMMA C tiles). K loop: 16 steps over F + 1 step over conv features.
// B-slab is double-buffered and filled with GLOBAL_LOAD_ASYNC_TO_LDS_B128
// (ASYNCcnt), so the fill of step s+1 overlaps the 8 WMMAs of step s and
// only ONE barrier per K-step is needed.
// ---------------------------------------------------------------------------
__launch_bounds__(256)
__global__ void attn_scores(const float*    __restrict__ kv,
                            const float*    __restrict__ weight,
                            const float*    __restrict__ Wconv,
                            const float*    __restrict__ qv,     // q+bias
                            const float*    __restrict__ Ws,
                            const _Float16* __restrict__ BT,
                            float*          __restrict__ scores) {
    __shared__ _Float16 slab[2][32 * 136]; // B tile: 32(K) x 128(H), rows padded to 136
    __shared__ _Float16 cfs[128 * 40];     // conv features: 128(t) x 32(c), rows padded to 40
    __shared__ float    wsl[160];          // weight slice with +/-15 halo
    __shared__ float    wcv[C_ * KC_];     // conv weights
    __shared__ float    qb[H_];            // q[b] + bias
    __shared__ float    wsr[H_];           // Ws

    const int tid  = threadIdx.x;
    const int wave = tid >> 5, lane = tid & 31;
    const int half = lane >> 4, l15 = lane & 15;
    const int b    = blockIdx.x >> 4;          // 16 T-tiles per batch
    const int t0   = (blockIdx.x & 15) * 128;

    // per-thread slab-fill coordinates (32B per thread, coalesced)
    const int fk  = tid >> 3;
    const int fch = (tid & 7) * 16;

    // async fill of one 32x128 f16 B-slab: LDS offset in %0, 64-bit global
    // address in %1; per ISA §10.7 the instruction offset is added to BOTH
    // the LDS and global addresses, so offset:16 moves the second chunk.
    #define SLAB_FILL_ASYNC(sstep, bufsel)                                      \
        do {                                                                    \
            unsigned _lds = (unsigned)(size_t)(&slab[(bufsel)][fk * 136 + fch]);\
            unsigned long long _ga =                                            \
                (unsigned long long)(BT + (size_t)((sstep) * 32 + fk) * H_ + fch);\
            asm volatile(                                                       \
                "global_load_async_to_lds_b128 %0, %1, off\n\t"                 \
                "global_load_async_to_lds_b128 %0, %1, off offset:16"           \
                :: "v"(_lds), "v"(_ga) : "memory");                             \
        } while (0)

    // kick off the first B-slab fill immediately (overlaps staging below)
    SLAB_FILL_ASYNC(0, 0);

    // ---- stage small operands ----
    for (int i = tid; i < 158; i += 256) {
        int g = t0 + i - 15;                   // conv "same" padding halo
        wsl[i] = (g >= 0 && g < T_) ? weight[b * T_ + g] : 0.0f;
    }
    for (int i = tid; i < C_ * KC_; i += 256) wcv[i] = Wconv[i];
    if (tid < H_) { qb[tid] = qv[b * H_ + tid]; wsr[tid] = Ws[tid]; }
    __syncthreads();

    // ---- conv features cf[t][c] (the K-augmentation slice of A) ----
    for (int o = tid; o < 128 * C_; o += 256) {
        int t = o >> 5, c = o & 31;
        float s = 0.f;
        #pragma unroll
        for (int k = 0; k < KC_; ++k) s += wcv[c * KC_ + k] * wsl[t + k];
        cfs[t * 40 + c] = (_Float16)s;
    }
    asm volatile("s_wait_asynccnt 0" ::: "memory");  // slab 0 resident
    __syncthreads();

    // ---- main GEMM: 8 C tiles per wave (16 rows x 128 cols) ----
    v8f acc[8];
    #pragma unroll
    for (int n = 0; n < 8; ++n)
        #pragma unroll
        for (int e = 0; e < 8; ++e) acc[n][e] = 0.0f;

    const int rowA = wave * 16 + l15;          // local row this lane feeds (A: lane=M)

    for (int s = 0; s < NKSTEP; ++s) {
        // prefetch next slab into the other buffer; overlaps the WMMAs below
        if (s + 1 < NKSTEP) SLAB_FILL_ASYNC(s + 1, (s + 1) & 1);

        // A fragment (16x32 f16): lane = M row; elems 0..7 -> K = half*8+i,
        // elems 8..15 -> K = 16 + half*8 + i   (ISA §7.12.2 16-bit A layout)
        v16h a;
        if (s < 16) {
            const float* ap = kv + ((size_t)(b * T_ + t0 + rowA)) * F_ + s * 32 + half * 8;
            if (s < 15) __builtin_prefetch(ap + 32, 0, 0);   // next K-step's kv
            float fv[16];
            *(float4*)(&fv[0])  = *(const float4*)(ap);
            *(float4*)(&fv[4])  = *(const float4*)(ap + 4);
            *(float4*)(&fv[8])  = *(const float4*)(ap + 16);
            *(float4*)(&fv[12]) = *(const float4*)(ap + 20);
            #pragma unroll
            for (int i = 0; i < 16; ++i) a[i] = (_Float16)fv[i];
        } else {                                // augmented slice: conv features
            const _Float16* cp = &cfs[rowA * 40];
            #pragma unroll
            for (int i = 0; i < 8; ++i) {
                a[i]     = cp[half * 8 + i];
                a[8 + i] = cp[16 + half * 8 + i];
            }
        }

        // B fragments (32x16 f16): lane = N col (l15), elems 0..7 -> K = half*8+i, elems 8..15 -> K = 16+half*8+i
        const _Float16* srow = &slab[s & 1][l15];
        #pragma unroll
        for (int n = 0; n < 8; ++n) {
            const _Float16* bp = srow + n * 16;
            v16h bb;
            #pragma unroll
            for (int i = 0; i < 8; ++i) { bb[i] = bp[(half * 8 + i) * 136]; bb[8 + i] = bp[(16 + half * 8 + i) * 136]; }
            acc[n] = __builtin_amdgcn_wmma_f32_16x16x32_f16(
                false, a, false, bb, (short)0, acc[n], false, false);
        }

        // next buffer must be resident before anyone reads it next iteration
        asm volatile("s_wait_asynccnt 0" ::: "memory");
        __syncthreads();
    }
    #undef SLAB_FILL_ASYNC

    // ---- epilogue: tanh + Ws-dot, reduce 16 lanes (one wave32 half) per row
    // C layout: VGPR j -> M = half*8 + j ; N = l15  (ISA §7.12.2 32-bit C/D)
    #pragma unroll
    for (int j = 0; j < 8; ++j) {
        int tl = wave * 16 + half * 8 + j;
        float p = 0.f;
        #pragma unroll
        for (int n = 0; n < 8; ++n) {
            int h = n * 16 + l15;
            float e = tanhf(acc[n][j] + qb[h]);
            p += wsr[h] * e;
        }
        p += __shfl_xor(p, 1, 32);
        p += __shfl_xor(p, 2, 32);
        p += __shfl_xor(p, 4, 32);
        p += __shfl_xor(p, 8, 32);
        if (l15 == 0) scores[b * T_ + t0 + tl] = p;
    }
}

// ---------------------------------------------------------------------------
// Kernel 4: masked softmax over T per batch row.
// ---------------------------------------------------------------------------
__launch_bounds__(256)
__global__ void softmax_t(const float* __restrict__ scores,
                          const unsigned char* __restrict__ mask,
                          float* __restrict__ attn) {
    __shared__ float red[8];
    const int b = blockIdx.x, tid = threadIdx.x;
    float v[8];
    float mx = -__builtin_inff();
    #pragma unroll
    for (int i = 0; i < 8; ++i) {
        int t = i * 256 + tid;
        float s = mask[b * T_ + t] ? scores[b * T_ + t] : -__builtin_inff();
        v[i] = s; mx = fmaxf(mx, s);
    }
    for (int m = 1; m < 32; m <<= 1) mx = fmaxf(mx, __shfl_xor(mx, m, 32));
    if ((tid & 31) == 0) red[tid >> 5] = mx;
    __syncthreads();
    mx = red[0];
    #pragma unroll
    for (int i = 1; i < 8; ++i) mx = fmaxf(mx, red[i]);
    __syncthreads();

    float sm = 0.f;
    #pragma unroll
    for (int i = 0; i < 8; ++i) {
        float e = (v[i] == -__builtin_inff()) ? 0.f : __expf(v[i] - mx);
        v[i] = e; sm += e;
    }
    for (int m = 1; m < 32; m <<= 1) sm += __shfl_xor(sm, m, 32);
    if ((tid & 31) == 0) red[tid >> 5] = sm;
    __syncthreads();
    sm = 0.f;
    #pragma unroll
    for (int i = 0; i < 8; ++i) sm += red[i];
    float inv = 1.0f / sm;
    #pragma unroll
    for (int i = 0; i < 8; ++i) attn[b * T_ + i * 256 + tid] = v[i] * inv;
}

// ---------------------------------------------------------------------------
// Kernels 5/6: context = sum_t attn[b,t] * kv[b,t,f], split 8-ways over T
// with a deterministic two-stage reduce (no fp atomics -> replay-stable).
// ---------------------------------------------------------------------------
__launch_bounds__(256)
__global__ void ctx_partial(const float* __restrict__ kv,
                            const float* __restrict__ attn,
                            float* __restrict__ partial) {
    __shared__ float al[256];
    const int idx = blockIdx.x;              // B_*16
    const int b = idx >> 4;
    const int r = idx & 15;
    const int fh = r & 1, sp = r >> 1;       // f-half (0/1), T-split (0..7)
    const int tid = threadIdx.x;
    al[tid] = attn[b * T_ + sp * 256 + tid];
    __syncthreads();
    const int f = fh * 256 + tid;
    const float* kp = kv + ((size_t)(b * T_ + sp * 256)) * F_ + f;
    float acc = 0.f;
    for (int tt = 0; tt < 256; ++tt) acc += al[tt] * kp[(size_t)tt * F_];
    partial[(size_t)(b * 8 + sp) * F_ + f] = acc;
}

__global__ void ctx_reduce(const float* __restrict__ partial,
                           float* __restrict__ out) {
    int o = blockIdx.x * 256 + threadIdx.x;  // B_*F_
    int b = o >> 9, f = o & 511;
    float s = 0.f;
    #pragma unroll
    for (int sp = 0; sp < 8; ++sp) s += partial[(size_t)(b * 8 + sp) * F_ + f];
    out[o] = s;
}

// ---------------------------------------------------------------------------
extern "C" void kernel_launch(void* const* d_in, const int* in_sizes, int n_in,
                              void* d_out, int out_size, void* d_ws, size_t ws_size,
                              hipStream_t stream) {
    const float*         query  = (const float*)d_in[0];
    const float*         kv     = (const float*)d_in[1];
    const unsigned char* mask   = (const unsigned char*)d_in[2];
    const float*         weight = (const float*)d_in[3];
    const float*         Wq     = (const float*)d_in[4];
    const float*         Wk     = (const float*)d_in[5];
    const float*         Ww     = (const float*)d_in[6];
    const float*         Wconv  = (const float*)d_in[7];
    const float*         bias   = (const float*)d_in[8];
    const float*         Ws     = (const float*)d_in[9];

    char* ws = (char*)d_ws;
    float*    q_ws    = (float*)(ws);                               // 16 KB
    _Float16* BT      = (_Float16*)(ws + 16384);                    // 136 KB
    float*    scores  = (float*)(ws + 16384 + 139264);              // 256 KB
    float*    partial = (float*)(ws + 16384 + 139264 + 262144);     // 512 KB

    float* ctx  = (float*)d_out;                 // [B,F]
    float* attn = (float*)d_out + B_ * F_;       // [B,T]

    build_bt   <<<(KAUG * H_ + 255) / 256, 256, 0, stream>>>(Wk, Ww, BT);
    qproj      <<<(B_ * H_) / 256,         256, 0, stream>>>(query, Wq, bias, q_ws);
    attn_scores<<<B_ * (T_ / 128),         256, 0, stream>>>(kv, weight, Wconv, q_ws, Ws, BT, scores);
    softmax_t  <<<B_,                      256, 0, stream>>>(scores, mask, attn);
    ctx_partial<<<B_ * 16,                 256, 0, stream>>>(kv, attn, partial);
    ctx_reduce <<<(B_ * F_) / 256,         256, 0, stream>>>(partial, ctx);
}
